// Net_5695126635082
// MI455X (gfx1250) — matrix operand order repaired and checked by reading, not compiled, not run on hardware
//
#include <hip/hip_runtime.h>

// ---------- types ----------
typedef __attribute__((ext_vector_type(16))) __bf16 v16bf;
typedef __attribute__((ext_vector_type(8)))  float  v8f;

// ---------- bf16 helpers ----------
__device__ inline __bf16 f2bf(float f) {
    union { float f; unsigned u; } v; v.f = f;
    unsigned r = v.u + 0x7FFFu + ((v.u >> 16) & 1u);   // round-to-nearest-even
    unsigned short h = (unsigned short)(r >> 16);
    __bf16 o; __builtin_memcpy(&o, &h, 2); return o;
}
__device__ inline float bf2f(__bf16 b) {
    unsigned short h; __builtin_memcpy(&h, &b, 2);
    union { unsigned u; float f; } v; v.u = ((unsigned)h) << 16; return v.f;
}

// ---------- gfx1250 async global->LDS (ASYNCcnt path) ----------
// LDS offset of a __shared__ object = low 32 bits of its generic address
// (see ISA flat addressing).
__device__ inline unsigned lds_off_of(const void* p) {
    return (unsigned)(unsigned long long)p;
}
__device__ inline void async_g2l_b128(unsigned lds_byte_off, const void* gaddr) {
    asm volatile("global_load_async_to_lds_b128 %0, %1, off"
                 :: "v"(lds_byte_off), "v"(gaddr) : "memory");
}
__device__ inline void wait_async0() {
    asm volatile("s_wait_asynccnt 0" ::: "memory");
}

// ---------- WMMA fragment loaders (CDNA5 ISA 7.12.2 layouts, wave32) ----------
// A (16x32 bf16): lanes 0-15 hold row M=lane, K = {0..7,16..23}; lanes 16-31 same row, K+8.
__device__ inline v16bf load_b_frag(const __bf16* Bm, int ldn, int lane) {
    // B (32x16 bf16): lane = n + 16*g holds column n, K = 8*g + i for i<8, 16 + 8*g + (i-8) for i>=8
    int n  = lane & 15;
    int kb = (lane >> 4) << 3;
    v16bf v;
#pragma unroll
    for (int i = 0; i < 16; ++i) v[i] = Bm[(kb + i + ((i >> 3) << 3)) * ldn + n];
    return v;
}
// C/D (16x16 f32): lane = n + 16*g, acc[j] -> (M = j + 8*g, N = n)

// ---------- head params ----------
struct HeadPs { const float *w1, *b1, *w2, *b2, *w3, *b3, *w4, *b4; };
struct FcParams { HeadPs h[5]; };

// ws layout (bytes)
#define WS_BC1   0          // 96*16 bf16
#define WS_BC2   3072       // 160*16 bf16
#define WS_WFC   8192       // 416*64 bf16
#define WS_BFC   61440      // 64 f32
#define WS_W4    61696      // 5*10*6 f32
#define WS_B4    62896      // 5*6 f32
#define WS_FEAT  65536      // B*416 bf16

// =====================================================================
// Kernel 1: build conv weight B-matrices, zero Wfc, bias chains, w4/b4 pack
// =====================================================================
__global__ void prep_small(const float* __restrict__ c1w, const float* __restrict__ c2w,
                           FcParams P, __bf16* Bc1, __bf16* Bc2, __bf16* Wfc,
                           float* biasFc, float* w4all, float* b4all) {
    const int tid = blockIdx.x * blockDim.x + threadIdx.x;
    const int stride = gridDim.x * blockDim.x;
    const __bf16 zb = f2bf(0.f);
    for (int i = tid; i < 96 * 16; i += stride) {       // conv1 (K=96, N=16)
        int k = i >> 4, n = i & 15;
        Bc1[i] = (k < 75 && n < 6) ? f2bf(c1w[n * 75 + k]) : zb;
    }
    for (int i = tid; i < 160 * 16; i += stride) {      // conv2 (K=160, N=16)
        int k = i >> 4, n = i & 15;
        Bc2[i] = (k < 150) ? f2bf(c2w[n * 150 + k]) : zb;
    }
    for (int i = tid; i < 416 * 64; i += stride) Wfc[i] = zb;
    const int W[5] = {4, 3, 5, 6, 5};                   // head widths
    for (int i = tid; i < 5 * 10 * 6; i += stride) {
        int h = i / 60, r = i % 60, ii = r / 6, j = r % 6;
        w4all[i] = (j < W[h]) ? P.h[h].w4[ii * W[h] + j] : 0.f;
    }
    for (int i = tid; i < 30; i += stride) {
        int h = i / 6, j = i % 6;
        b4all[i] = (j < W[h]) ? P.h[h].b4[j] : 0.f;
    }
    if (tid < 5) {    // bias chain ((b1@w2)+b2)@w3+b3 -> biasFc[10h..10h+9]
        const HeadPs hp = P.h[tid];
        float t[84];
        for (int j = 0; j < 84; ++j) {
            float a = hp.b2[j];
            for (int k = 0; k < 120; ++k) a += hp.b1[k] * hp.w2[k * 84 + j];
            t[j] = a;
        }
        for (int j = 0; j < 10; ++j) {
            float a = hp.b3[j];
            for (int m = 0; m < 84; ++m) a += t[m] * hp.w3[m * 10 + j];
            biasFc[tid * 10 + j] = a;
        }
    }
    if (tid >= 50 && tid < 64) biasFc[tid] = 0.f;
}

// =====================================================================
// Kernel 2: compose each head's 400x10 linear stack into Wfc columns
// =====================================================================
__global__ void compose_kernel(FcParams P, __bf16* Wfc) {
    int gid = blockIdx.x * blockDim.x + threadIdx.x;
    if (gid >= 2000) return;
    int hd = gid / 400, r = gid % 400;
    const HeadPs hp = P.h[hd];
    float t1[84];
    for (int j = 0; j < 84; ++j) t1[j] = 0.f;
    for (int k = 0; k < 120; ++k) {
        float a = hp.w1[r * 120 + k];
        for (int j = 0; j < 84; ++j) t1[j] += a * hp.w2[k * 84 + j];
    }
    for (int j = 0; j < 10; ++j) {
        float a = 0.f;
        for (int m = 0; m < 84; ++m) a += t1[m] * hp.w3[m * 10 + j];
        Wfc[r * 64 + hd * 10 + j] = f2bf(a);
    }
}

// =====================================================================
// Kernel 3: fused conv1+relu+pool+conv2+relu+pool per image, WMMA bf16
// one block per image, 4 waves. Image staged via async global->LDS DMA.
// Padded im2col offsets point into zeroed LDS regions -> no predication.
// =====================================================================
__global__ __launch_bounds__(128) void conv_feat_kernel(
        const float* __restrict__ x, const float* __restrict__ c1b,
        const float* __restrict__ c2b, const __bf16* __restrict__ Bc1,
        const __bf16* __restrict__ Bc2, __bf16* __restrict__ feat) {
    // imgf is dead after the bf16 conversion barrier; overlay it with c2buf
    __shared__ union __align__(16) {
        float imgf[3 * 1024];        // raw f32 image (async DMA dst)
        float c2buf[112 * 16];       // conv2 relu out (f32)
    } u;
    __shared__ __bf16 img[3 * 1024 + 1024];           // bf16 image + zero pad region
    __shared__ __bf16 c1buf[784 * 8];                 // conv1 relu out, ch padded to 8
    __shared__ __bf16 p1buf[6 * 196 + 200];           // pooled conv1 + zero pad region
    __shared__ int    off1[96];                       // im2col offsets conv1
    __shared__ int    off2[160];                      // im2col offsets conv2
    __shared__ __bf16 Bc1s[96 * 16];
    __shared__ __bf16 Bc2s[160 * 16];

    const int tid = threadIdx.x, lane = tid & 31, wave = tid >> 5;
    const int b = blockIdx.x;
    const __bf16 zb = f2bf(0.f);

    // ---- kick off async image DMA (768 x 16B), overlap with weight staging ----
    {
        const char* gsrc = (const char*)(x + (size_t)b * 3072);
        const unsigned lbase = lds_off_of(u.imgf);
#pragma unroll
        for (int j = 0; j < 6; ++j) {
            const int idx = tid + j * 128;            // 16-byte chunk index
            async_g2l_b128(lbase + idx * 16, gsrc + idx * 16);
        }
    }
    // im2col offset tables: padded K entries point at zeroed LDS scratch
    for (int k = tid; k < 96; k += 128)
        off1[k] = (k < 75) ? ((k / 25) * 1024 + ((k % 25) / 5) * 32 + (k % 5)) : 3072;
    for (int k = tid; k < 160; k += 128)
        off2[k] = (k < 150) ? ((k / 25) * 196 + ((k % 25) / 5) * 14 + (k % 5)) : 1176;
    for (int i = tid; i < 1024; i += 128) img[3072 + i] = zb;   // zero region (base <= 891)
    for (int i = tid; i < 200; i += 128) p1buf[1176 + i] = zb;  // zero region (base <= 195)
    for (int i = tid; i < 96 * 16; i += 128) Bc1s[i] = Bc1[i];
    for (int i = tid; i < 160 * 16; i += 128) Bc2s[i] = Bc2[i];
    wait_async0();
    __syncthreads();
    // f32 -> bf16 conversion pass
    for (int i = tid; i < 3072; i += 128) img[i] = f2bf(u.imgf[i]);
    __syncthreads();

    const int n = lane & 15, hi = lane >> 4;
    const int klo = hi << 3;
    // hoist loop-invariant B fragments and channel biases into registers
    v16bf bW1[3], bW2[5];
#pragma unroll
    for (int s = 0; s < 3; ++s) bW1[s] = load_b_frag(Bc1s + s * 32 * 16, 16, lane);
#pragma unroll
    for (int s = 0; s < 5; ++s) bW2[s] = load_b_frag(Bc2s + s * 32 * 16, 16, lane);
    const float bias1 = (n < 6) ? c1b[n] : 0.f;
    const float bias2 = c2b[n];

    // ---- conv1: M=784 (49 tiles), N=16 (6 live), K=96 (3 wmma steps) ----
    for (int t = wave; t < 49; t += 4) {
        v8f acc = {};
        const int p = t * 16 + n;
        const int base = (p / 28) * 32 + (p % 28);
#pragma unroll
        for (int s = 0; s < 3; ++s) {
            const int kb = s * 32 + klo;
            v16bf a;
#pragma unroll
            for (int i = 0; i < 8; ++i) {
                a[i]     = img[off1[kb + i] + base];          // unconditional gathers
                a[8 + i] = img[off1[kb + 16 + i] + base];
            }
            acc = __builtin_amdgcn_wmma_f32_16x16x32_bf16(false, a, false, bW1[s],
                                                          (short)0, acc, false, false);
        }
        if (n < 6) {
#pragma unroll
            for (int j = 0; j < 8; ++j) {
                int pp = t * 16 + j + 8 * hi;
                float v = acc[j] + bias1;
                c1buf[pp * 8 + n] = f2bf(v > 0.f ? v : 0.f);
            }
        }
    }
    __syncthreads();

    // ---- 2x2 maxpool -> p1buf (6,14,14) ----
    for (int i = tid; i < 6 * 196; i += 128) {
        int c = i / 196, r = i % 196, py = r / 14, px = r % 14;
        float m = -1e30f;
#pragma unroll
        for (int dy = 0; dy < 2; ++dy)
#pragma unroll
            for (int dx = 0; dx < 2; ++dx) {
                int p = (2 * py + dy) * 28 + (2 * px + dx);
                float v = bf2f(c1buf[p * 8 + c]);
                m = v > m ? v : m;
            }
        p1buf[c * 196 + py * 14 + px] = f2bf(m);
    }
    __syncthreads();

    // ---- conv2: M=100 (7 tiles, last padded), N=16, K=160 (5 wmma steps) ----
    for (int t = wave; t < 7; t += 4) {
        v8f acc = {};
        int p = t * 16 + n; if (p > 99) p = 99;
        const int base = (p / 10) * 14 + (p % 10);
#pragma unroll
        for (int s = 0; s < 5; ++s) {
            const int kb = s * 32 + klo;
            v16bf a;
#pragma unroll
            for (int i = 0; i < 8; ++i) {
                a[i]     = p1buf[off2[kb + i] + base];
                a[8 + i] = p1buf[off2[kb + 16 + i] + base];
            }
            acc = __builtin_amdgcn_wmma_f32_16x16x32_bf16(false, a, false, bW2[s],
                                                          (short)0, acc, false, false);
        }
#pragma unroll
        for (int j = 0; j < 8; ++j) {
            int pp = t * 16 + j + 8 * hi;
            if (pp < 100) {
                float v = acc[j] + bias2;
                u.c2buf[pp * 16 + n] = v > 0.f ? v : 0.f;
            }
        }
    }
    __syncthreads();

    // ---- 2x2 maxpool -> feat row (400 live + 16 zero pad), bf16 ----
    __bf16* frow = feat + (size_t)b * 416;
    for (int i = tid; i < 400; i += 128) {
        int c = i / 25, r = i % 25, py = r / 5, px = r % 5;
        float m = -1e30f;
#pragma unroll
        for (int dy = 0; dy < 2; ++dy)
#pragma unroll
            for (int dx = 0; dx < 2; ++dx) {
                float v = u.c2buf[((2 * py + dy) * 10 + (2 * px + dx)) * 16 + c];
                m = v > m ? v : m;
            }
        frow[i] = f2bf(m);
    }
    if (tid < 16) frow[400 + tid] = zb;
}

// =====================================================================
// Kernel 4: feat(B,416) @ Wfc(416,64) via WMMA + routed-MoE epilogue
// block = 128 threads (4 waves), 64 rows per block
// =====================================================================
__global__ __launch_bounds__(128) void fc_kernel(
        const __bf16* __restrict__ feat, const __bf16* __restrict__ Wfc,
        const float* __restrict__ biasFc, const float* __restrict__ w4all,
        const float* __restrict__ b4all, float* __restrict__ out, int B) {
    __shared__ __align__(16) __bf16 Ws[416 * 64];
    __shared__ float zbuf[4][16][64];
    __shared__ float bfs[64];

    const int tid = threadIdx.x, lane = tid & 31, wave = tid >> 5;
    {   // stage Wfc into LDS (16B-wide copies)
        const uint4* src = (const uint4*)Wfc;
        uint4* dst = (uint4*)Ws;
        for (int i = tid; i < (416 * 64 * 2) / 16; i += 128) dst[i] = src[i];
        if (tid < 64) bfs[tid] = biasFc[tid];
    }
    __syncthreads();

    const int r0 = blockIdx.x * 64 + wave * 16;
    const __bf16* A = feat + (size_t)r0 * 416;
    v8f acc0 = {}, acc1 = {}, acc2 = {}, acc3 = {};
    const int arow = lane & 15;
    for (int s = 0; s < 13; ++s) {
        const __bf16* ap = A + (size_t)arow * 416 + s * 32 + ((lane >> 4) << 3);
        if (s + 1 < 13) __builtin_prefetch(ap + 32, 0, 0);
        v16bf a;
#pragma unroll
        for (int i = 0; i < 8; ++i) { a[i] = ap[i]; a[8 + i] = ap[16 + i]; }
        const __bf16* wb = Ws + s * 32 * 64;
        v16bf b0 = load_b_frag(wb +  0, 64, lane);
        v16bf b1 = load_b_frag(wb + 16, 64, lane);
        v16bf b2 = load_b_frag(wb + 32, 64, lane);
        v16bf b3 = load_b_frag(wb + 48, 64, lane);
        acc0 = __builtin_amdgcn_wmma_f32_16x16x32_bf16(false, a, false, b0, (short)0, acc0, false, false);
        acc1 = __builtin_amdgcn_wmma_f32_16x16x32_bf16(false, a, false, b1, (short)0, acc1, false, false);
        acc2 = __builtin_amdgcn_wmma_f32_16x16x32_bf16(false, a, false, b2, (short)0, acc2, false, false);
        acc3 = __builtin_amdgcn_wmma_f32_16x16x32_bf16(false, a, false, b3, (short)0, acc3, false, false);
    }
    {   // spill z = feat@Wfc + bias to LDS
        int n = lane & 15, hi = lane >> 4;
#pragma unroll
        for (int j = 0; j < 8; ++j) {
            int m = j + 8 * hi;
            zbuf[wave][m][ 0 + n] = acc0[j] + bfs[ 0 + n];
            zbuf[wave][m][16 + n] = acc1[j] + bfs[16 + n];
            zbuf[wave][m][32 + n] = acc2[j] + bfs[32 + n];
            zbuf[wave][m][48 + n] = acc3[j] + bfs[48 + n];
        }
    }
    __syncthreads();

    // ---- epilogue: one row per thread (tid < 64) ----
    if (tid < 64) {
        const int wv = tid >> 4, m = tid & 15;
        const int row = blockIdx.x * 64 + tid;
        float z[64];
#pragma unroll
        for (int i = 0; i < 64; ++i) z[i] = zbuf[wv][m][i];
        float s4[4];
#pragma unroll
        for (int j = 0; j < 4; ++j) {
            float a = b4all[j];
            for (int i = 0; i < 10; ++i) {
                float zi = z[i] > 0.f ? z[i] : 0.f;
                a += zi * w4all[i * 6 + j];
            }
            s4[j] = a;
        }
        int e = 0; float best = s4[0];
#pragma unroll
        for (int j = 1; j < 4; ++j) if (s4[j] > best) { best = s4[j]; e = j; }
#pragma unroll
        for (int j = 0; j < 4; ++j) out[(size_t)row * 4 + j] = s4[j];
        const int widths[4] = {3, 5, 6, 5};
        const int h = 1 + e;
        float* orow = out + (size_t)B * 4 + (size_t)row * 6;
#pragma unroll
        for (int j = 0; j < 6; ++j) {
            float a = 0.f;
            if (j < widths[e]) {
                a = b4all[h * 6 + j];
                for (int i = 0; i < 10; ++i) a += z[10 * h + i] * w4all[(h * 10 + i) * 6 + j];
            }
            orow[j] = a;
        }
    }
}

// =====================================================================
// launch
// =====================================================================
static void fill_heads(void* const* d_in, int n_in, FcParams& P) {
    static const int W[5] = {4, 3, 5, 6, 5};
    if (n_in >= 45) {   // pytree leaves passed individually
        for (int h = 0; h < 5; ++h) {
            int b = 5 + 8 * h;
            P.h[h].w1 = (const float*)d_in[b + 0];
            P.h[h].b1 = (const float*)d_in[b + 1];
            P.h[h].w2 = (const float*)d_in[b + 2];
            P.h[h].b2 = (const float*)d_in[b + 3];
            P.h[h].w3 = (const float*)d_in[b + 4];
            P.h[h].b3 = (const float*)d_in[b + 5];
            P.h[h].w4 = (const float*)d_in[b + 6];
            P.h[h].b4 = (const float*)d_in[b + 7];
        }
    } else {            // each tuple flattened into one blob
        for (int h = 0; h < 5; ++h) {
            const float* base = (const float*)d_in[5 + h];
            size_t o = 0;
            P.h[h].w1 = base + o; o += 400 * 120;
            P.h[h].b1 = base + o; o += 120;
            P.h[h].w2 = base + o; o += 120 * 84;
            P.h[h].b2 = base + o; o += 84;
            P.h[h].w3 = base + o; o += 84 * 10;
            P.h[h].b3 = base + o; o += 10;
            P.h[h].w4 = base + o; o += 10 * (size_t)W[h];
            P.h[h].b4 = base + o;
        }
    }
}

extern "C" void kernel_launch(void* const* d_in, const int* in_sizes, int n_in,
                              void* d_out, int out_size, void* d_ws, size_t ws_size,
                              hipStream_t stream) {
    const float* x   = (const float*)d_in[0];
    const float* c1w = (const float*)d_in[1];
    const float* c1b = (const float*)d_in[2];
    const float* c2w = (const float*)d_in[3];
    const float* c2b = (const float*)d_in[4];
    FcParams P;
    fill_heads(d_in, n_in, P);

    char* w = (char*)d_ws;
    __bf16* Bc1    = (__bf16*)(w + WS_BC1);
    __bf16* Bc2    = (__bf16*)(w + WS_BC2);
    __bf16* Wfc    = (__bf16*)(w + WS_WFC);
    float*  biasFc = (float*)(w + WS_BFC);
    float*  w4all  = (float*)(w + WS_W4);
    float*  b4all  = (float*)(w + WS_B4);
    __bf16* FEAT   = (__bf16*)(w + WS_FEAT);

    const int B = in_sizes[0] / (3 * 32 * 32);   // 16384

    prep_small<<<26, 256, 0, stream>>>(c1w, c2w, P, Bc1, Bc2, Wfc, biasFc, w4all, b4all);
    compose_kernel<<<8, 256, 0, stream>>>(P, Wfc);
    conv_feat_kernel<<<B, 128, 0, stream>>>(x, c1b, c2b, Bc1, Bc2, FEAT);
    fc_kernel<<<B / 64, 128, 0, stream>>>(FEAT, Wfc, biasFc, w4all, b4all, (float*)d_out, B);
    (void)out_size; (void)ws_size; (void)in_sizes;
}
